// GenericMoeLayer_53094385713155
// MI455X (gfx1250) — matrix operand order repaired and checked by reading, not compiled, not run on hardware
//
#include <hip/hip_runtime.h>
#include <hip/hip_bf16.h>
#include <cstdint>

// ---------------- problem constants ----------------
#define T_TOK   8192
#define HDIM    1024
#define NEXP    64
#define FDIM    768     // per-expert FFN width
#define TWO_F   1536
#define SFDIM   4096    // shared expert width
#define TOPK    8
#define MT      64      // token tile (rows) per FFN block
#define FC      512     // shared-expert F chunk

typedef __attribute__((ext_vector_type(16))) __bf16 bf16x16;
typedef __attribute__((ext_vector_type(8)))  __bf16 bf16x8;
typedef __attribute__((ext_vector_type(8)))  float  f32x8;

// ---------------- WMMA helper (CDNA5 bf16, 16x16x32, fp32 accum) ----------------
static __device__ __forceinline__ f32x8 wmma_bf16(bf16x16 a, bf16x16 b, f32x8 c) {
  return __builtin_amdgcn_wmma_f32_16x16x32_bf16(false, a, false, b, (short)0, c,
                                                 false, false);
}

// A fragment: 16x32 bf16 tile from LDS (row-major, stride ld elements).
// Contiguous k order used here (same as B, not the ISA table order): lanes 0-15 row M=lane, K = 0..15 ; lanes 16-31 same rows,
// K = 16..31  -> 16 contiguous elements (two 16B loads) per lane.
static __device__ __forceinline__ void load_a_lds(const __bf16* base, int ld,
                                                  int m0, int kb, bf16x16& a) {
  const int lane = threadIdx.x & 31;
  const __bf16* p = base + (size_t)(m0 + (lane & 15)) * ld + kb + ((lane >> 4) << 4);
  bf16x8 lo = *(const bf16x8*)(p);        // ds_load_b128
  bf16x8 hi = *(const bf16x8*)(p + 8);    // ds_load_b128
#pragma unroll
  for (int i = 0; i < 8; ++i) { a[i] = lo[i]; a[i + 8] = hi[i]; }
}

// B fragment, bf16 weights: element (k,n) = W[n0+n, kb+k], W row-major stride ldw.
// Lane n holds column n; lanes 0-15 K=0..15, lanes 16-31 K=16..31
// -> 16 contiguous bf16 per lane = two global_load_b128, zero VALU conversion.
static __device__ __forceinline__ void load_b_frag(const __bf16* __restrict__ W,
                                                   int ldw, int n0, int kb,
                                                   bf16x16& b) {
  const int lane = threadIdx.x & 31;
  const __bf16* p = W + (size_t)(n0 + (lane & 15)) * ldw + kb + ((lane >> 4) << 4);
  __builtin_prefetch(p + 128, 0, 1);      // global_prefetch_b8 (next K chunks)
  bf16x8 lo = *(const bf16x8*)(p);
  bf16x8 hi = *(const bf16x8*)(p + 8);
#pragma unroll
  for (int i = 0; i < 8; ++i) { b[i] = lo[i]; b[i + 8] = hi[i]; }
}

// B fragment, fp32 weights (fallback when ws too small to hold bf16 copies).
static __device__ __forceinline__ void load_b_frag(const float* __restrict__ W,
                                                   int ldw, int n0, int kb,
                                                   bf16x16& b) {
  const int lane = threadIdx.x & 31;
  const float* p = W + (size_t)(n0 + (lane & 15)) * ldw + kb + ((lane >> 4) << 4);
  __builtin_prefetch(p + 64, 0, 1);
  const float4* q = (const float4*)p;
  float4 f0 = q[0], f1 = q[1], f2 = q[2], f3 = q[3];
  b[0]  = (__bf16)f0.x; b[1]  = (__bf16)f0.y; b[2]  = (__bf16)f0.z; b[3]  = (__bf16)f0.w;
  b[4]  = (__bf16)f1.x; b[5]  = (__bf16)f1.y; b[6]  = (__bf16)f1.z; b[7]  = (__bf16)f1.w;
  b[8]  = (__bf16)f2.x; b[9]  = (__bf16)f2.y; b[10] = (__bf16)f2.z; b[11] = (__bf16)f2.w;
  b[12] = (__bf16)f3.x; b[13] = (__bf16)f3.y; b[14] = (__bf16)f3.z; b[15] = (__bf16)f3.w;
}

// Fast sigmoid / silu: v_exp_f32 + v_rcp_f32 (no precise-division expansion).
static __device__ __forceinline__ float fast_sigmoid(float g) {
  return __builtin_amdgcn_rcpf(1.0f + __expf(-g));
}
static __device__ __forceinline__ float silu_f(float g) {
  return g * fast_sigmoid(g);
}

// ---------------- zero / convert helpers ----------------
__global__ void moe_zero_f32(float* __restrict__ p, int n) {
  int i = blockIdx.x * blockDim.x + threadIdx.x;
  if (i < n) p[i] = 0.0f;
}
__global__ void moe_zero_i32(int* __restrict__ p, int n) {
  int i = blockIdx.x * blockDim.x + threadIdx.x;
  if (i < n) p[i] = 0;
}
// fp32 -> bf16 bulk conversion, 8 elements / thread (n8 = n/8)
__global__ void moe_cvt_bf16(const float* __restrict__ src, __bf16* __restrict__ dst,
                             int n8) {
  int i = blockIdx.x * blockDim.x + threadIdx.x;
  if (i >= n8) return;
  const float4* s = (const float4*)src + (size_t)i * 2;
  float4 a = s[0], c = s[1];
  bf16x8 o;
  o[0] = (__bf16)a.x; o[1] = (__bf16)a.y; o[2] = (__bf16)a.z; o[3] = (__bf16)a.w;
  o[4] = (__bf16)c.x; o[5] = (__bf16)c.y; o[6] = (__bf16)c.z; o[7] = (__bf16)c.w;
  *(bf16x8*)(dst + (size_t)i * 8) = o;
}

// ---------------- router (one 64-thread block per token) ----------------
__global__ __launch_bounds__(64)
void moe_router(const float* __restrict__ hidden, const float* __restrict__ gate_w,
                const float* __restrict__ shared_gate_w,
                int* __restrict__ topk_id, float* __restrict__ topk_w,
                float* __restrict__ gate_out, int* __restrict__ counts) {
  __shared__ float xs[HDIM];
  __shared__ float logits[NEXP];
  __shared__ float red[NEXP];
  const int t = blockIdx.x;
  const int tid = threadIdx.x;
  const float* x = hidden + (size_t)t * HDIM;
  for (int i = tid; i < HDIM / 4; i += 64)
    ((float4*)xs)[i] = ((const float4*)x)[i];
  __syncthreads();

  const float* gw = gate_w + (size_t)tid * HDIM;
  float acc = 0.0f;
  for (int k = 0; k < HDIM; ++k) acc = fmaf(xs[k], gw[k], acc);
  logits[tid] = acc;

  float gacc = 0.0f;
  for (int k = tid; k < HDIM; k += 64) gacc = fmaf(xs[k], shared_gate_w[k], gacc);
  red[tid] = gacc;
  __syncthreads();

  if (tid == 0) {
    float gs = 0.0f;
    for (int i = 0; i < NEXP; ++i) gs += red[i];
    gate_out[t] = fast_sigmoid(gs);

    float mx = logits[0];
    for (int i = 1; i < NEXP; ++i) mx = fmaxf(mx, logits[i]);
    float p[NEXP];
    float sum = 0.0f;
    for (int i = 0; i < NEXP; ++i) { p[i] = __expf(logits[i] - mx); sum += p[i]; }
    const float inv = __builtin_amdgcn_rcpf(sum);

    int ids[TOPK]; float ws[TOPK]; float wsum = 0.0f;
    for (int j = 0; j < TOPK; ++j) {
      int bi = 0; float bv = -1.0f;
      for (int i = 0; i < NEXP; ++i)
        if (p[i] > bv) { bv = p[i]; bi = i; }
      p[bi] = -2.0f;
      ids[j] = bi; ws[j] = bv * inv; wsum += ws[j];
    }
    const float rn = __builtin_amdgcn_rcpf(wsum);
    for (int j = 0; j < TOPK; ++j) {
      topk_id[t * TOPK + j] = ids[j];
      topk_w[t * TOPK + j]  = ws[j] * rn;
      atomicAdd(&counts[ids[j]], 1);
    }
  }
}

// ---------------- scan + scatter ----------------
__global__ void moe_scan(const int* __restrict__ counts, int* __restrict__ offsets,
                         int* __restrict__ cursors) {
  if (threadIdx.x == 0) {
    int acc = 0;
    for (int e = 0; e < NEXP; ++e) { offsets[e] = acc; acc += counts[e]; }
    offsets[NEXP] = acc;
  }
  if (threadIdx.x < NEXP) cursors[threadIdx.x] = 0;
}

__global__ void moe_scatter(const int* __restrict__ topk_id,
                            const float* __restrict__ topk_w,
                            const int* __restrict__ offsets, int* __restrict__ cursors,
                            int* __restrict__ tok_list, float* __restrict__ w_list) {
  int i = blockIdx.x * blockDim.x + threadIdx.x;
  if (i >= T_TOK * TOPK) return;
  int t = i >> 3;
  int e = topk_id[i];
  int p = offsets[e] + atomicAdd(&cursors[e], 1);
  tok_list[p] = t;
  w_list[p]   = topk_w[i];
}

// ---------------- grouped expert FFN (gather + fused siglu) ----------------
// grid = (NEXP, maxTiles); block = 256 (8 waves); dynamic LDS ~232 KB.
template <typename WT>
__global__ __launch_bounds__(256)
void moe_expert_ffn(const float* __restrict__ hidden, const WT* __restrict__ w1,
                    const WT* __restrict__ w2, const int* __restrict__ counts,
                    const int* __restrict__ offsets, const int* __restrict__ tok_list,
                    const float* __restrict__ w_list, float* __restrict__ out) {
  const int XS = HDIM + 8;   // 1032 bf16 elems
  const int AS = FDIM + 8;   // 776 bf16 elems
  extern __shared__ char smem[];
  __bf16* x_lds   = (__bf16*)smem;                               // [MT][XS]
  __bf16* act_lds = (__bf16*)(smem + (size_t)MT * XS * 2);       // [MT][AS]
  int*    tok_s   = (int*)  (smem + (size_t)MT * (XS + AS) * 2); // [MT]
  float*  ws_s    = (float*)(tok_s + MT);                        // [MT]

  const int e    = blockIdx.x;
  const int tile = blockIdx.y;
  const int cnt  = counts[e];
  if (tile * MT >= cnt) return;
  const int base = offsets[e];
  const int tid  = threadIdx.x;
  const int lane = tid & 31, wave = tid >> 5;

  if (tid < MT) {
    int r  = tile * MT + tid;
    int rr = (r < cnt) ? r : (cnt - 1);
    tok_s[tid] = tok_list[base + rr];
    ws_s[tid]  = (r < cnt) ? w_list[base + rr] : 0.0f;
  }
  __syncthreads();

  for (int i4 = tid; i4 < MT * (HDIM / 4); i4 += 256) {
    int row = i4 >> 8, c4 = (i4 & 255) << 2;
    float4 v = *(const float4*)(hidden + (size_t)tok_s[row] * HDIM + c4);
    __bf16* d = x_lds + (size_t)row * XS + c4;
    d[0] = (__bf16)v.x; d[1] = (__bf16)v.y; d[2] = (__bf16)v.z; d[3] = (__bf16)v.w;
  }
  __syncthreads();

  const WT* w1e = w1 + (size_t)e * TWO_F * HDIM;  // rows [0,F)=gate, [F,2F)=up
  const WT* w2e = w2 + (size_t)e * HDIM * FDIM;

  // Phase 1: h = x @ w1^T ; act = silu(g)*u  (tile-pairs share the A fragment)
  for (int tp = wave; tp < (MT / 16) * (FDIM / 16); tp += 8) {
    const int mt = tp / (FDIM / 16), ft = tp % (FDIM / 16);
    f32x8 cg = {}; f32x8 cu = {};
#pragma unroll 2
    for (int kb = 0; kb < HDIM; kb += 32) {
      bf16x16 a, bg, bu;
      load_a_lds(x_lds, XS, mt * 16, kb, a);
      load_b_frag(w1e, HDIM, ft * 16, kb, bg);
      load_b_frag(w1e + (size_t)FDIM * HDIM, HDIM, ft * 16, kb, bu);
      cg = wmma_bf16(a, bg, cg);
      cu = wmma_bf16(a, bu, cu);
    }
    const int col = lane & 15, half = lane >> 4;
#pragma unroll
    for (int v = 0; v < 8; ++v) {
      int m = mt * 16 + v + 8 * half;   // C/D layout: vgpr v -> row v (+8 hi lanes)
      act_lds[(size_t)m * AS + ft * 16 + col] = (__bf16)(silu_f(cg[v]) * cu[v]);
    }
  }
  __syncthreads();

  // Phase 2: y = act @ w2^T, scaled atomic accumulate into out
  for (int tt = wave; tt < (MT / 16) * (HDIM / 16); tt += 8) {
    const int mt = tt / (HDIM / 16), jt = tt % (HDIM / 16);
    f32x8 c = {};
#pragma unroll 2
    for (int kb = 0; kb < FDIM; kb += 32) {
      bf16x16 a, b;
      load_a_lds(act_lds, AS, mt * 16, kb, a);
      load_b_frag(w2e, FDIM, jt * 16, kb, b);
      c = wmma_bf16(a, b, c);
    }
    const int col = lane & 15, half = lane >> 4;
#pragma unroll
    for (int v = 0; v < 8; ++v) {
      int m = mt * 16 + v + 8 * half;
      float wgt = ws_s[m];
      if (wgt != 0.0f)
        unsafeAtomicAdd(out + (size_t)tok_s[m] * HDIM + jt * 16 + col, c[v] * wgt);
    }
  }
}

// ---------------- shared expert (chunked over SF) ----------------
// grid = (T/MT, SFDIM/FC); block = 256; dynamic LDS ~199 KB.
template <typename WT>
__global__ __launch_bounds__(256)
void moe_shared_ffn(const float* __restrict__ hidden, const WT* __restrict__ sw1,
                    const WT* __restrict__ sw2, const float* __restrict__ gate,
                    float* __restrict__ out) {
  const int XS = HDIM + 8;   // 1032
  const int AS = FC + 8;     // 520
  extern __shared__ char smem[];
  __bf16* x_lds   = (__bf16*)smem;                          // [MT][XS]
  __bf16* act_lds = (__bf16*)(smem + (size_t)MT * XS * 2);  // [MT][AS]

  const int t0    = blockIdx.x * MT;
  const int chunk = blockIdx.y;          // f in [chunk*FC, chunk*FC+FC)
  const int tid   = threadIdx.x;
  const int lane  = tid & 31, wave = tid >> 5;

  for (int i4 = tid; i4 < MT * (HDIM / 4); i4 += 256) {
    int row = i4 >> 8, c4 = (i4 & 255) << 2;
    float4 v = *(const float4*)(hidden + (size_t)(t0 + row) * HDIM + c4);
    __bf16* d = x_lds + (size_t)row * XS + c4;
    d[0] = (__bf16)v.x; d[1] = (__bf16)v.y; d[2] = (__bf16)v.z; d[3] = (__bf16)v.w;
  }
  __syncthreads();

  const WT* w1g = sw1 + (size_t)(chunk * FC) * HDIM;           // gate rows
  const WT* w1u = sw1 + (size_t)(SFDIM + chunk * FC) * HDIM;   // up rows

  // Phase 1
  for (int tp = wave; tp < (MT / 16) * (FC / 16); tp += 8) {
    const int mt = tp / (FC / 16), ft = tp % (FC / 16);
    f32x8 cg = {}; f32x8 cu = {};
#pragma unroll 2
    for (int kb = 0; kb < HDIM; kb += 32) {
      bf16x16 a, bg, bu;
      load_a_lds(x_lds, XS, mt * 16, kb, a);
      load_b_frag(w1g, HDIM, ft * 16, kb, bg);
      load_b_frag(w1u, HDIM, ft * 16, kb, bu);
      cg = wmma_bf16(a, bg, cg);
      cu = wmma_bf16(a, bu, cu);
    }
    const int col = lane & 15, half = lane >> 4;
#pragma unroll
    for (int v = 0; v < 8; ++v) {
      int m = mt * 16 + v + 8 * half;
      act_lds[(size_t)m * AS + ft * 16 + col] = (__bf16)(silu_f(cg[v]) * cu[v]);
    }
  }
  __syncthreads();

  // Phase 2: y[m,j] = sum_f act[m,f] * sw2[j, chunk*FC + f]
  for (int tt = wave; tt < (MT / 16) * (HDIM / 16); tt += 8) {
    const int mt = tt / (HDIM / 16), jt = tt % (HDIM / 16);
    f32x8 c = {};
#pragma unroll 2
    for (int kb = 0; kb < FC; kb += 32) {
      bf16x16 a, b;
      load_a_lds(act_lds, AS, mt * 16, kb, a);
      load_b_frag(sw2, SFDIM, jt * 16, chunk * FC + kb, b);
      c = wmma_bf16(a, b, c);
    }
    const int col = lane & 15, half = lane >> 4;
#pragma unroll
    for (int v = 0; v < 8; ++v) {
      int m  = mt * 16 + v + 8 * half;
      int tg = t0 + m;
      unsafeAtomicAdd(out + (size_t)tg * HDIM + jt * 16 + col, c[v] * gate[tg]);
    }
  }
}

// ---------------- host-side orchestration ----------------
// workspace layout (bytes):
//   [0 .. 2MB): router/scatter plumbing
//   [2MB ..  ): optional bf16 weight copies (guarded by ws_size)
#define WS_SMALL   (2u * 1024u * 1024u)
#define W1_ELEMS   ((size_t)NEXP * TWO_F * HDIM)      // 100,663,296
#define W2_ELEMS   ((size_t)NEXP * HDIM * FDIM)       //  50,331,648
#define SW1_ELEMS  ((size_t)2 * SFDIM * HDIM)         //   8,388,608
#define SW2_ELEMS  ((size_t)HDIM * SFDIM)             //   4,194,304

extern "C" void kernel_launch(void* const* d_in, const int* in_sizes, int n_in,
                              void* d_out, int out_size, void* d_ws, size_t ws_size,
                              hipStream_t stream) {
  (void)in_sizes; (void)n_in; (void)out_size;
  const float* hidden  = (const float*)d_in[0];   // [T, H]
  const float* gate_w  = (const float*)d_in[1];   // [E, H]
  const float* w1      = (const float*)d_in[2];   // [E, 2F, H]
  const float* w2      = (const float*)d_in[3];   // [E, H, F]
  const float* sw1     = (const float*)d_in[4];   // [2SF, H]
  const float* sw2     = (const float*)d_in[5];   // [H, SF]
  const float* sgate_w = (const float*)d_in[6];   // [1, H]
  float* out = (float*)d_out;                     // [T, H]

  char* ws = (char*)d_ws;
  int*   topk_id  = (int*)  (ws + 0);
  float* topk_w   = (float*)(ws + 262144);
  int*   tok_list = (int*)  (ws + 524288);
  float* w_list   = (float*)(ws + 786432);
  float* gate     = (float*)(ws + 1048576);
  int*   counts   = (int*)  (ws + 1081344);
  int*   cursors  = (int*)  (ws + 1081600);
  int*   offsets  = (int*)  (ws + 1081856);

  // 1) routing plumbing
  moe_zero_i32<<<1, 128, 0, stream>>>(counts, 64);
  moe_zero_i32<<<1, 128, 0, stream>>>(cursors, 64);
  moe_router<<<T_TOK, 64, 0, stream>>>(hidden, gate_w, sgate_w,
                                       topk_id, topk_w, gate, counts);
  moe_scan<<<1, 64, 0, stream>>>(counts, offsets, cursors);
  moe_scatter<<<(T_TOK * TOPK + 255) / 256, 256, 0, stream>>>(
      topk_id, topk_w, offsets, cursors, tok_list, w_list);

  // 2) zero output (atomic accumulation target)
  moe_zero_f32<<<(T_TOK * HDIM + 255) / 256, 256, 0, stream>>>(out, T_TOK * HDIM);

  const size_t expert_lds = (size_t)MT * (HDIM + 8) * 2 + (size_t)MT * (FDIM + 8) * 2 +
                            MT * (sizeof(int) + sizeof(float));
  const size_t shared_lds = (size_t)MT * (HDIM + 8) * 2 + (size_t)MT * (FC + 8) * 2;
  dim3 egrid(NEXP, T_TOK / MT);       // worst case: all tokens on one expert
  dim3 sgrid(T_TOK / MT, SFDIM / FC);

  const size_t bf16_need =
      WS_SMALL + 2 * (W1_ELEMS + W2_ELEMS + SW1_ELEMS + SW2_ELEMS);

  if (ws_size >= bf16_need) {
    // 3a) pre-convert all weights to bf16 (halves HBM/L2 bytes on the re-streamed
    //     operand, deletes per-fragment cvt chains from the WMMA hot loop)
    __bf16* w1b  = (__bf16*)(ws + WS_SMALL);
    __bf16* w2b  = w1b + W1_ELEMS;
    __bf16* sw1b = w2b + W2_ELEMS;
    __bf16* sw2b = sw1b + SW1_ELEMS;
    moe_cvt_bf16<<<(int)((W1_ELEMS / 8 + 255) / 256), 256, 0, stream>>>(w1, w1b,
                                                           (int)(W1_ELEMS / 8));
    moe_cvt_bf16<<<(int)((W2_ELEMS / 8 + 255) / 256), 256, 0, stream>>>(w2, w2b,
                                                           (int)(W2_ELEMS / 8));
    moe_cvt_bf16<<<(int)((SW1_ELEMS / 8 + 255) / 256), 256, 0, stream>>>(sw1, sw1b,
                                                           (int)(SW1_ELEMS / 8));
    moe_cvt_bf16<<<(int)((SW2_ELEMS / 8 + 255) / 256), 256, 0, stream>>>(sw2, sw2b,
                                                           (int)(SW2_ELEMS / 8));
    moe_expert_ffn<__bf16><<<egrid, 256, expert_lds, stream>>>(
        hidden, w1b, w2b, counts, offsets, tok_list, w_list, out);
    moe_shared_ffn<__bf16><<<sgrid, 256, shared_lds, stream>>>(
        hidden, sw1b, sw2b, gate, out);
  } else {
    // 3b) fallback: stream fp32 weights, convert per fragment
    moe_expert_ffn<float><<<egrid, 256, expert_lds, stream>>>(
        hidden, w1, w2, counts, offsets, tok_list, w_list, out);
    moe_shared_ffn<float><<<sgrid, 256, shared_lds, stream>>>(
        hidden, sw1, sw2, gate, out);
  }
}
